// RNNEncoder_25022479467074
// MI455X (gfx1250) — compile-verified, matrix operand order checked by reading, not run on hardware
//
#include <hip/hip_runtime.h>

#define S_LEN 512
#define B_SZ  64
#define D_IN  512
#define H_SZ  512
#define G4H   2048   // 4*H

typedef __bf16 bf16_t;
typedef __attribute__((ext_vector_type(16))) __bf16       v16bf;
typedef __attribute__((ext_vector_type(8)))  float        v8f;
typedef __attribute__((ext_vector_type(8)))  unsigned int v8u;
typedef __attribute__((ext_vector_type(4)))  int          v4i;
typedef __attribute__((ext_vector_type(4)))  unsigned int v4ui;
typedef __attribute__((ext_vector_type(8)))  int          v8i;
typedef __attribute__((ext_vector_type(4)))  int          v4si;

// Dynamic-LDS layout for the recurrent kernel (fits in 320KB WGP LDS):
//   Bs: 16 tiles * 16 kchunks * (32 lanes * 16 elems) bf16 = 256 KB  (Whh slice, fragment-ordered)
//   As: 16 rows * 512 K bf16                               =  16 KB  (h tile staging)
//   Gs: 4 gates * 16 rows * 64 cols f32                    =  16 KB  (activated gates)
#define REC_SMEM_B 262144
#define REC_SMEM_A 16384
#define REC_SMEM_G 16384
#define REC_SMEM_TOTAL (REC_SMEM_B + REC_SMEM_A + REC_SMEM_G)  // 294912 B

#ifndef __has_builtin
#define __has_builtin(x) 0
#endif
#if __has_builtin(__builtin_amdgcn_tensor_load_to_lds)
#define USE_TDM 1
#else
#define USE_TDM 0
#endif

static __device__ __forceinline__ float sigmoidf_(float x) {
    return 1.0f / (1.0f + __expf(-x));
}

#if USE_TDM
// ---------------------------------------------------------------------------
// Issue a 2D TENSOR_LOAD_TO_LDS (TDM DMA). Descriptor per CDNA5 ISA ch.8:
//   group0: count=1, lds_addr, global_addr[56:0], type=2 ("image")
//   group1: data_size=3 (8B), tensor/tile dims, dim0 stride (8B units)
// tile_d0: row length in 8-byte units (<= 65535); tile_d1: #rows;
// row_stride8: distance between rows in 8-byte units.
// Tracked by TENSORcnt; caller waits with s_wait_tensorcnt.
// ---------------------------------------------------------------------------
template <int CPOL>
static __device__ __forceinline__ void tdm_load_2d(unsigned lds_addr,
                                                   const void* gptr,
                                                   unsigned tile_d0,
                                                   unsigned tile_d1,
                                                   unsigned long long row_stride8) {
    unsigned long long ga = (unsigned long long)gptr;
    v4ui g0;
    g0[0] = 1u;                                      // count=1, user-mode D#
    g0[1] = lds_addr;                                // LDS byte address
    g0[2] = (unsigned)(ga & 0xffffffffu);            // global_addr[31:0]
    g0[3] = (unsigned)((ga >> 32) & 0x01ffffffu) |   // global_addr[56:32]
            0x80000000u;                             // type=2 ("image")

    v8i g1;
    g1[0] = (int)(3u << 16);                         // data_size=3 (8B), mask=0
    g1[1] = (int)((tile_d0 & 0xffffu) << 16);        // tensor_dim0[15:0]
    g1[2] = (int)(((tile_d0 >> 16) & 0xffffu) |      // tensor_dim0[31:16]
                  ((tile_d1 & 0xffffu) << 16));      // tensor_dim1[15:0]
    g1[3] = (int)(((tile_d1 >> 16) & 0xffffu) |      // tensor_dim1[31:16]
                  ((tile_d0 & 0xffffu) << 16));      // tile_dim0
    g1[4] = (int)(tile_d1 & 0xffffu);                // tile_dim1; tile_dim2=0
    g1[5] = (int)(row_stride8 & 0xffffffffu);        // tensor_dim0_stride[31:0]
    g1[6] = (int)((row_stride8 >> 32) & 0xffffu);    // stride[47:32]; dim1_stride=0
    g1[7] = 0;

    v4si gz = {0, 0, 0, 0};
#if __clang_major__ >= 23
    v8i gz8 = {0, 0, 0, 0, 0, 0, 0, 0};
    __builtin_amdgcn_tensor_load_to_lds(g0, g1, gz, gz, gz8, CPOL);
#else
    __builtin_amdgcn_tensor_load_to_lds(g0, g1, gz, gz, CPOL);
#endif
}
#endif  // USE_TDM

// ---------------------------------------------------------------------------
// f32 -> bf16 elementwise conversion
// ---------------------------------------------------------------------------
__global__ void cvt_f32_bf16_kernel(const float* __restrict__ src,
                                    bf16_t* __restrict__ dst, int n) {
    for (int i = blockIdx.x * blockDim.x + threadIdx.x; i < n;
         i += gridDim.x * blockDim.x)
        dst[i] = (bf16_t)src[i];
}

// ---------------------------------------------------------------------------
// Pack a weight matrix W (N x K, row-major f32) into WMMA B-fragment order:
//   P[ntile][kchunk][lane][i]  (bf16), where B[k][n] = W[n][k] and the
//   per-lane element order matches the 16-bit 16x32 operand layout:
//     n = ntile*16 + (lane&15), khalf = lane>>4
//     i<8 : k = kchunk*32 + khalf*8 + i
//     i>=8: k = kchunk*32 + 16 + khalf*8 + (i-8)
// So each lane's 16 bf16 (32B) are contiguous -> fragment loads are 2x b128.
// ---------------------------------------------------------------------------
__global__ void pack_wfrag_kernel(const float* __restrict__ W,
                                  bf16_t* __restrict__ P, int N, int K) {
    const int kchunks = K >> 5;
    const int total = N * K;
    for (int idx = blockIdx.x * blockDim.x + threadIdx.x; idx < total;
         idx += gridDim.x * blockDim.x) {
        int i      = idx & 15;
        int lane   = (idx >> 4) & 31;
        int rest   = idx >> 9;
        int kchunk = rest % kchunks;
        int ntile  = rest / kchunks;
        int n      = ntile * 16 + (lane & 15);
        int khalf  = lane >> 4;
        int klocal = (i < 8) ? (khalf * 8 + i) : (16 + khalf * 8 + (i - 8));
        int k      = kchunk * 32 + klocal;
        P[idx] = (bf16_t)W[(size_t)n * K + k];
    }
}

// ---------------------------------------------------------------------------
// Input-projection GEMM: G[dir][m][n] = X[m][:] . Wih[n][:] + bih[n] + bhh[n]
//   X: (M x K) bf16 row-major, WP*: fragment-packed bf16, G: f32.
// Block = 256 threads (8 waves). Block tile 64(M) x 64(N), wave tile 16x32.
// A tile is double-buffered in LDS: one barrier per K-chunk, global loads for
// chunk kc+1 overlap the WMMAs of chunk kc.
// ---------------------------------------------------------------------------
__global__ __launch_bounds__(256) void wih_gemm_kernel(
    const bf16_t* __restrict__ X,
    const bf16_t* __restrict__ WPf, const bf16_t* __restrict__ WPb,
    const float* __restrict__ bihf, const float* __restrict__ bhhf,
    const float* __restrict__ bihb, const float* __restrict__ bhhb,
    float* __restrict__ G, int M, int K) {
    __shared__ bf16_t As[2][64][32];

    const int dir = blockIdx.z;
    const bf16_t* __restrict__ WP  = dir ? WPb : WPf;
    const float* __restrict__ bih  = dir ? bihb : bihf;
    const float* __restrict__ bhh  = dir ? bhhb : bhhf;
    float* __restrict__ Gd = G + (size_t)dir * (size_t)M * G4H;

    const int tid   = threadIdx.x;
    const int wave  = tid >> 5;
    const int lane  = tid & 31;
    const int wm    = wave & 3;   // m-tile within block
    const int wn    = wave >> 2;  // n-tile pair: {2wn, 2wn+1}
    const int row16 = lane & 15;
    const int khalf = lane >> 4;
    const int mblock  = blockIdx.x;
    const int nblock  = blockIdx.y;
    const int kchunks = K >> 5;

    v8f acc0 = {0.f, 0.f, 0.f, 0.f, 0.f, 0.f, 0.f, 0.f};
    v8f acc1 = {0.f, 0.f, 0.f, 0.f, 0.f, 0.f, 0.f, 0.f};

    auto stageA = [&](int kc, int buf) {
        int r = tid >> 2, seg = tid & 3;
        const v4i* src =
            (const v4i*)(X + (size_t)(mblock * 64 + r) * K + kc * 32 + seg * 8);
        *(v4i*)(&As[buf][r][seg * 8]) = *src;
    };

    stageA(0, 0);
    __syncthreads();

    for (int kc = 0; kc < kchunks; ++kc) {
        const int cur = kc & 1;
        if (kc + 1 < kchunks) stageA(kc + 1, cur ^ 1);

        // A fragment per the 16-bit 16x32 layout (u32 holds a K-pair).
        const unsigned int* Arow =
            (const unsigned int*)(&As[cur][wm * 16 + row16][0]);
        const int a0 = khalf * 4;
        const int a1 = 8 + khalf * 4;
        v8u araw;
        araw[0] = Arow[a0];     araw[1] = Arow[a0 + 1];
        araw[2] = Arow[a0 + 2]; araw[3] = Arow[a0 + 3];
        araw[4] = Arow[a1];     araw[5] = Arow[a1 + 1];
        araw[6] = Arow[a1 + 2]; araw[7] = Arow[a1 + 3];
        v16bf a = __builtin_bit_cast(v16bf, araw);

        // B fragments straight from packed global (32B/lane, coalesced).
        const int nt0 = nblock * 4 + 2 * wn;
        v8u b0raw = *(const v8u*)(WP + ((size_t)nt0 * kchunks + kc) * 512 + lane * 16);
        v8u b1raw = *(const v8u*)(WP + ((size_t)(nt0 + 1) * kchunks + kc) * 512 + lane * 16);
        v16bf b0 = __builtin_bit_cast(v16bf, b0raw);
        v16bf b1 = __builtin_bit_cast(v16bf, b1raw);

        acc0 = __builtin_amdgcn_wmma_f32_16x16x32_bf16(false, a, false, b0,
                                                       (short)0, acc0, false, false);
        acc1 = __builtin_amdgcn_wmma_f32_16x16x32_bf16(false, a, false, b1,
                                                       (short)0, acc1, false, false);
        __syncthreads();
    }

    // Epilogue: add both biases, store f32 pre-gates.
#pragma unroll
    for (int t = 0; t < 2; ++t) {
        v8f acc = t ? acc1 : acc0;
        int n = (nblock * 4 + 2 * wn + t) * 16 + row16;
        float bias = bih[n] + bhh[n];
        int mbase = mblock * 64 + wm * 16 + khalf * 8;
#pragma unroll
        for (int r = 0; r < 8; ++r)
            Gd[(size_t)(mbase + r) * G4H + n] = acc[r] + bias;
    }
}

// ---------------------------------------------------------------------------
// Persistent recurrent kernel. 64 WGs: dir(2) x batch-tile(4) x hcol-tile(8).
// Whh slice (256 gate-cols x 512 K) lives in LDS for all 512 steps, loaded by
// ONE tensor_load_to_lds DMA (4 rows of 64KB, 512KB apart = the 4 gates'
// contiguous tile groups). Per step the 16x512 h tile is staged by another
// 2D TDM op (16 rows x 1KB, 2KB stride), then h @ Whh-slice -> 16x256 gates
// via WMMA; activations; cell update; inter-WG barrier among the 8 hcol-tiles
// sharing (dir, batch-tile).
// ---------------------------------------------------------------------------
__global__ __launch_bounds__(256) void lstm_recurrent_kernel(
    const float* __restrict__ G,       // [dir][S*B][2048] pre-gates (+bias)
    const bf16_t* __restrict__ WPf, const bf16_t* __restrict__ WPb,
    bf16_t* __restrict__ Hout,         // (S,B,1024) bf16 h (both dirs)
    float* __restrict__ OutF32,        // layer-1: d_out h section; else null
    const float* __restrict__ mask,    // (S,B)
    float* __restrict__ hn, float* __restrict__ cn,  // (B,1024) this layer
    int* __restrict__ syncc) {
    extern __shared__ char smem[];
    bf16_t* Bs = (bf16_t*)smem;
    bf16_t* As = (bf16_t*)(smem + REC_SMEM_B);
    float*  Gs = (float*)(smem + REC_SMEM_B + REC_SMEM_A);

    const int bx  = blockIdx.x;
    const int dir = bx >> 5;
    const int bt  = (bx >> 3) & 3;
    const int jc  = bx & 7;
    const int grp = bx >> 3;  // (dir,bt) group: 8 WGs per group

    const bf16_t* __restrict__ WP = dir ? WPb : WPf;
    const float* __restrict__ Gd = G + (size_t)dir * (size_t)(S_LEN * B_SZ) * G4H;

    const int tid   = threadIdx.x;
    const int wave  = tid >> 5;
    const int lane  = tid & 31;
    const int row16 = lane & 15;
    const int khalf = lane >> 4;
    const int p  = wave >> 2;  // gate pair: 0 -> {i,f}, 1 -> {g,o}
    const int cs = wave & 3;   // 16-col sub-block within the 64-col slice

    // One-time Whh-slice preload into LDS (256 KB).
#if USE_TDM
    if (wave == 0) {
        // 4 gate groups (rows), each 4 contiguous 16-col tiles = 64KB;
        // row stride = 32 tiles = 512KB = 65536 x 8B.
        tdm_load_2d<0>((unsigned)(size_t)(void*)Bs,
                       WP + (size_t)(jc * 4) * 8192,
                       /*tile_d0=*/8192, /*tile_d1=*/4,
                       /*row_stride8=*/65536ull);
        __builtin_amdgcn_s_wait_tensorcnt(0);
    }
#else
    for (int t = 0; t < 16; ++t) {
        int gate = t >> 2, csb = t & 3;
        int ntg = gate * 32 + jc * 4 + csb;
        const v4i* src = (const v4i*)(WP + (size_t)ntg * 8192);
        v4i* dst = (v4i*)(Bs + t * 8192);
        for (int idx = tid; idx < 1024; idx += 256) dst[idx] = src[idx];
    }
#endif

    // Elementwise-update ownership: 4 cells per thread.
    const int colE   = tid & 63;
    const int mbaseE = (tid >> 6) * 4;
    float c_st[4] = {0.f, 0.f, 0.f, 0.f};
    float h_st[4] = {0.f, 0.f, 0.f, 0.f};
    __syncthreads();

    for (int it = 0; it < S_LEN; ++it) {
        const int s = dir ? (S_LEN - 1 - it) : it;
        v8f acc0 = {0.f, 0.f, 0.f, 0.f, 0.f, 0.f, 0.f, 0.f};
        v8f acc1 = {0.f, 0.f, 0.f, 0.f, 0.f, 0.f, 0.f, 0.f};

        if (it > 0) {
            const int sp = dir ? (s + 1) : (s - 1);
            // Stage previous h tile (16 rows x 512 K bf16 = 16 KB) into LDS.
#if USE_TDM
            if (wave == 0) {
                const bf16_t* src =
                    Hout + ((size_t)(sp * B_SZ + bt * 16)) * 1024 + dir * 512;
                // 16 rows x 1KB, row stride 2KB; DEV scope (0x10) so the DMA
                // reads peers' h through L2, not a stale near cache.
                tdm_load_2d<0x10>((unsigned)(size_t)(void*)As, src,
                                  /*tile_d0=*/128, /*tile_d1=*/16,
                                  /*row_stride8=*/256ull);
                __builtin_amdgcn_s_wait_tensorcnt(0);
            }
#else
            {
                int r = tid >> 4, seg = tid & 15;
                const v4i* src = (const v4i*)(Hout +
                    ((size_t)(sp * B_SZ + bt * 16 + r)) * 1024 + dir * 512 + seg * 32);
                v4i* dst = (v4i*)(As + r * 512 + seg * 32);
                dst[0] = src[0]; dst[1] = src[1]; dst[2] = src[2]; dst[3] = src[3];
            }
#endif
            __syncthreads();
            const int t0 = (2 * p) * 4 + cs;
            const int t1 = (2 * p + 1) * 4 + cs;
            for (int kc = 0; kc < 16; ++kc) {
                const unsigned int* Arow =
                    (const unsigned int*)(As + row16 * 512 + kc * 32);
                const int a0 = khalf * 4;
                const int a1 = 8 + khalf * 4;
                v8u araw;
                araw[0] = Arow[a0];     araw[1] = Arow[a0 + 1];
                araw[2] = Arow[a0 + 2]; araw[3] = Arow[a0 + 3];
                araw[4] = Arow[a1];     araw[5] = Arow[a1 + 1];
                araw[6] = Arow[a1 + 2]; araw[7] = Arow[a1 + 3];
                v16bf a = __builtin_bit_cast(v16bf, araw);
                v8u b0raw = *(const v8u*)(Bs + t0 * 8192 + kc * 512 + lane * 16);
                v8u b1raw = *(const v8u*)(Bs + t1 * 8192 + kc * 512 + lane * 16);
                v16bf b0 = __builtin_bit_cast(v16bf, b0raw);
                v16bf b1 = __builtin_bit_cast(v16bf, b1raw);
                acc0 = __builtin_amdgcn_wmma_f32_16x16x32_bf16(
                    false, a, false, b0, (short)0, acc0, false, false);
                acc1 = __builtin_amdgcn_wmma_f32_16x16x32_bf16(
                    false, a, false, b1, (short)0, acc1, false, false);
            }
        }

        // Add precomputed pre-gates, activate, stash in LDS.
#pragma unroll
        for (int t = 0; t < 2; ++t) {
            int g = 2 * p + t;
            v8f acc = t ? acc1 : acc0;
            int ccol = cs * 16 + row16;          // 0..63 within slice
            int n = g * 512 + jc * 64 + ccol;    // global gate column
#pragma unroll
            for (int r = 0; r < 8; ++r) {
                int m = khalf * 8 + r;           // 0..15 batch row in tile
                int b = bt * 16 + m;
                float v = acc[r] + Gd[((size_t)s * B_SZ + b) * G4H + n];
                float act = (g == 2) ? tanhf(v) : sigmoidf_(v);
                Gs[(g * 16 + m) * 64 + ccol] = act;
            }
        }
        __syncthreads();

        // Cell update: c in registers, h broadcast via global bf16.
#pragma unroll
        for (int j = 0; j < 4; ++j) {
            int m = mbaseE + j;
            int b = bt * 16 + m;
            float iv = Gs[(0 * 16 + m) * 64 + colE];
            float fv = Gs[(1 * 16 + m) * 64 + colE];
            float gv = Gs[(2 * 16 + m) * 64 + colE];
            float ov = Gs[(3 * 16 + m) * 64 + colE];
            float c2 = fv * c_st[j] + iv * gv;
            float h2 = ov * tanhf(c2);
            float mt = mask[s * B_SZ + b];
            h2 *= mt;
            c2 *= mt;
            c_st[j] = c2;
            h_st[j] = h2;
            int col = jc * 64 + colE;
            size_t oidx = ((size_t)s * B_SZ + b) * 1024 + dir * 512 + col;
            Hout[oidx] = (bf16_t)h2;
            if (OutF32) OutF32[oidx] = h2;
        }

        // Warm L2 with next step's pre-gate rows while we wait on the barrier.
        if (it + 1 < S_LEN) {
            int sn = dir ? (s - 1) : (s + 1);
            const float* pf =
                &Gd[((size_t)sn * B_SZ + bt * 16 + (tid & 15)) * G4H + (tid >> 4) * 128];
            __builtin_prefetch(pf, 0, 1);
        }

        // Inter-WG step barrier among the 8 hcol-tiles of this (dir,bt).
        __threadfence();
        __syncthreads();
        if (tid == 0) {
            __hip_atomic_fetch_add(&syncc[grp * 32], 1, __ATOMIC_RELEASE,
                                   __HIP_MEMORY_SCOPE_AGENT);
            int target = 8 * (it + 1);
            while (__hip_atomic_load(&syncc[grp * 32], __ATOMIC_ACQUIRE,
                                     __HIP_MEMORY_SCOPE_AGENT) < target)
                __builtin_amdgcn_s_sleep(2);
        }
        __syncthreads();
    }

    // Final states hn, cn for this layer.
#pragma unroll
    for (int j = 0; j < 4; ++j) {
        int b = bt * 16 + mbaseE + j;
        int col = jc * 64 + colE;
        hn[(size_t)b * 1024 + dir * 512 + col] = h_st[j];
        cn[(size_t)b * 1024 + dir * 512 + col] = c_st[j];
    }
}

// ---------------------------------------------------------------------------
extern "C" void kernel_launch(void* const* d_in, const int* in_sizes, int n_in,
                              void* d_out, int out_size, void* d_ws,
                              size_t ws_size, hipStream_t stream) {
    (void)in_sizes; (void)n_in; (void)out_size; (void)ws_size;

    const float* x     = (const float*)d_in[0];
    const float* mask  = (const float*)d_in[1];
    const float* Wih0f = (const float*)d_in[2];
    const float* Whh0f = (const float*)d_in[3];
    const float* bih0f = (const float*)d_in[4];
    const float* bhh0f = (const float*)d_in[5];
    const float* Wih0b = (const float*)d_in[6];
    const float* Whh0b = (const float*)d_in[7];
    const float* bih0b = (const float*)d_in[8];
    const float* bhh0b = (const float*)d_in[9];
    const float* Wih1f = (const float*)d_in[10];
    const float* Whh1f = (const float*)d_in[11];
    const float* bih1f = (const float*)d_in[12];
    const float* bhh1f = (const float*)d_in[13];
    const float* Wih1b = (const float*)d_in[14];
    const float* Whh1b = (const float*)d_in[15];
    const float* bih1b = (const float*)d_in[16];
    const float* bhh1b = (const float*)d_in[17];

    float* out = (float*)d_out;
    char* ws = (char*)d_ws;

    size_t off = 0;
    auto take = [&](size_t bytes) {
        size_t r = off;
        off += (bytes + (size_t)255) & ~(size_t)255;
        return r;
    };
    int*    syncA  = (int*)(ws + take(4096));
    bf16_t* Xbf    = (bf16_t*)(ws + take((size_t)S_LEN * B_SZ * D_IN * 2));
    bf16_t* X1bf   = (bf16_t*)(ws + take((size_t)S_LEN * B_SZ * 2 * H_SZ * 2));
    bf16_t* H1buf  = (bf16_t*)(ws + take((size_t)S_LEN * B_SZ * 2 * H_SZ * 2));
    float*  Gbuf   = (float*)(ws + take((size_t)2 * S_LEN * B_SZ * G4H * 4));
    bf16_t* WihP0f = (bf16_t*)(ws + take((size_t)G4H * D_IN * 2));
    bf16_t* WihP0b = (bf16_t*)(ws + take((size_t)G4H * D_IN * 2));
    bf16_t* WihP1f = (bf16_t*)(ws + take((size_t)G4H * 2 * H_SZ * 2));
    bf16_t* WihP1b = (bf16_t*)(ws + take((size_t)G4H * 2 * H_SZ * 2));
    bf16_t* WhhP0f = (bf16_t*)(ws + take((size_t)G4H * H_SZ * 2));
    bf16_t* WhhP0b = (bf16_t*)(ws + take((size_t)G4H * H_SZ * 2));
    bf16_t* WhhP1f = (bf16_t*)(ws + take((size_t)G4H * H_SZ * 2));
    bf16_t* WhhP1b = (bf16_t*)(ws + take((size_t)G4H * H_SZ * 2));

    int* sync0 = syncA;        // 8 groups * 32-int stride
    int* sync1 = syncA + 256;

    float* hn = out + (size_t)S_LEN * B_SZ * 2 * H_SZ;   // (2, B, 2H)
    float* cn = hn + (size_t)2 * B_SZ * 2 * H_SZ;        // (2, B, 2H)

    (void)hipFuncSetAttribute((const void*)lstm_recurrent_kernel,
                              hipFuncAttributeMaxDynamicSharedMemorySize,
                              REC_SMEM_TOTAL);

    (void)hipMemsetAsync(syncA, 0, 4096, stream);

    // x -> bf16
    cvt_f32_bf16_kernel<<<4096, 256, 0, stream>>>(x, Xbf, S_LEN * B_SZ * D_IN);

    // Pack all weights into WMMA-fragment order (bf16).
    pack_wfrag_kernel<<<2048, 256, 0, stream>>>(Wih0f, WihP0f, G4H, D_IN);
    pack_wfrag_kernel<<<2048, 256, 0, stream>>>(Wih0b, WihP0b, G4H, D_IN);
    pack_wfrag_kernel<<<4096, 256, 0, stream>>>(Wih1f, WihP1f, G4H, 2 * H_SZ);
    pack_wfrag_kernel<<<4096, 256, 0, stream>>>(Wih1b, WihP1b, G4H, 2 * H_SZ);
    pack_wfrag_kernel<<<2048, 256, 0, stream>>>(Whh0f, WhhP0f, G4H, H_SZ);
    pack_wfrag_kernel<<<2048, 256, 0, stream>>>(Whh0b, WhhP0b, G4H, H_SZ);
    pack_wfrag_kernel<<<2048, 256, 0, stream>>>(Whh1f, WhhP1f, G4H, H_SZ);
    pack_wfrag_kernel<<<2048, 256, 0, stream>>>(Whh1b, WhhP1b, G4H, H_SZ);

    const int M = S_LEN * B_SZ;  // 32768

    // Layer 0: input GEMM (both dirs) then persistent recurrence.
    {
        dim3 grid(M / 64, G4H / 64, 2);
        wih_gemm_kernel<<<grid, 256, 0, stream>>>(Xbf, WihP0f, WihP0b, bih0f,
                                                  bhh0f, bih0b, bhh0b, Gbuf, M,
                                                  D_IN);
        lstm_recurrent_kernel<<<64, 256, REC_SMEM_TOTAL, stream>>>(
            Gbuf, WhhP0f, WhhP0b, X1bf, (float*)nullptr, mask,
            hn + 0 * (size_t)B_SZ * 1024, cn + 0 * (size_t)B_SZ * 1024, sync0);
    }

    // Layer 1: input GEMM consumes layer-0 bf16 output, recurrence writes d_out.
    {
        dim3 grid(M / 64, G4H / 64, 2);
        wih_gemm_kernel<<<grid, 256, 0, stream>>>(X1bf, WihP1f, WihP1b, bih1f,
                                                  bhh1f, bih1b, bhh1b, Gbuf, M,
                                                  2 * H_SZ);
        lstm_recurrent_kernel<<<64, 256, REC_SMEM_TOTAL, stream>>>(
            Gbuf, WhhP1f, WhhP1b, H1buf, out, mask,
            hn + 1 * (size_t)B_SZ * 1024, cn + 1 * (size_t)B_SZ * 1024, sync1);
    }
}
